// SlidingWindowAttention_4844723110072
// MI455X (gfx1250) — matrix operand order repaired and checked by reading, not compiled, not run on hardware
//
#include <hip/hip_runtime.h>
#include <hip/hip_bf16.h>

// ---------------------------------------------------------------------------
// Sliding-window attention, MI455X (gfx1250, wave32, WMMA).
// Mixed precision: bf16 operands, fp32 accumulation via v_wmma_f32_16x16x32_bf16.
// ---------------------------------------------------------------------------

typedef __bf16 bf16_t;
typedef __attribute__((ext_vector_type(16))) __bf16 v16bf;
typedef __attribute__((ext_vector_type(8)))  __bf16 v8bf;
typedef __attribute__((ext_vector_type(8)))  float  v8f;

#define EMBED 1024
#define NH    16
#define HD    64
#define SEQ   2048
#define NB    2
#define WIN   256

// ----- WMMA helpers --------------------------------------------------------

__device__ __forceinline__ v8f wmma_bf16(v16bf a, v16bf b, v8f c) {
  // (neg_a, A, neg_b, B, c_mod, C, reuse_a, reuse_b)
  return __builtin_amdgcn_wmma_f32_16x16x32_bf16(false, a, false, b, (short)0, c,
                                                 false, false);
}

// A-operand (16x32 bf16). Lane l (m = l&15): contiguous K range,
//   lanes 0-15 : h0..h15 = K 0..15   (kBase = 0)
//   lanes 16-31: h0..h15 = K 16..31  (kBase = 8)
// 'row' points at LDS row start of the 32-wide K chunk (16B aligned).
__device__ __forceinline__ v16bf make_a(const bf16_t* row, int kBase) {
  v8bf lo = *(const v8bf*)(row + 2 * kBase);
  v8bf hi = *(const v8bf*)(row + 2 * kBase + 8);
  v16bf a;
#pragma unroll
  for (int i = 0; i < 8; ++i) { a[i] = lo[i]; a[i + 8] = hi[i]; }
  return a;
}

// B-operand (32x16 bf16). Lane l (n = l&15): contiguous K range,
// lanes 0-15 K=0..15, lanes 16-31 K=16..31.  'p' points at LDS[n][kOff].
__device__ __forceinline__ v16bf make_b(const bf16_t* p) {
  v8bf lo = *(const v8bf*)(p);
  v8bf hi = *(const v8bf*)(p + 8);
  v16bf b;
#pragma unroll
  for (int i = 0; i < 8; ++i) { b[i] = lo[i]; b[i + 8] = hi[i]; }
  return b;
}

// ----- fp32 -> bf16 convert ------------------------------------------------

__global__ void __launch_bounds__(256)
cvt_f32_bf16(const float* __restrict__ in, bf16_t* __restrict__ out, int n4) {
  int i = blockIdx.x * blockDim.x + threadIdx.x;
  if (i < n4) {
    float4 v = ((const float4*)in)[i];
    bf16_t* o = out + 4 * (size_t)i;
    o[0] = (bf16_t)v.x; o[1] = (bf16_t)v.y; o[2] = (bf16_t)v.z; o[3] = (bf16_t)v.w;
  }
}

// ----- Generic bf16 GEMM: C[m][n] = sum_k A[m][k]*W[n][k] ------------------
// mode 0: out bf16 scattered to [B][H][S][D], val = (acc + bias[n]) * scale
// mode 1: out fp32 row-major [M][N],          val =  acc + bias[n]

__global__ void __launch_bounds__(128)
gemm_bf16(const bf16_t* __restrict__ A, const bf16_t* __restrict__ W,
          const float* __restrict__ bias, void* __restrict__ out,
          int M, int N, int K, int mode, float scale) {
  __shared__ bf16_t As[64][40];   // 64 rows x (32 + 8 pad), 80B rows (16B aligned)
  __shared__ bf16_t Bs[64][40];

  const int tid  = threadIdx.x;
  const int w    = tid >> 5;
  const int lane = tid & 31;
  const int wy   = w >> 1, wx = w & 1;     // 2x2 wave grid, 32x32 per wave
  const int m0   = blockIdx.y * 64;
  const int n0   = blockIdx.x * 64;
  const int nl   = lane & 15;
  const int half = lane >> 4;
  const int kBase = half * 8;
  const int kOff  = half * 16;

  v8f acc[2][2];
#pragma unroll
  for (int i = 0; i < 2; ++i)
#pragma unroll
    for (int j = 0; j < 2; ++j)
#pragma unroll
      for (int r = 0; r < 8; ++r) acc[i][j][r] = 0.f;

  const int lrow = tid >> 1;          // 0..63
  const int lcol = (tid & 1) * 16;    // 0 / 16

  for (int k0 = 0; k0 < K; k0 += 32) {
    __syncthreads();
    {
      const v8bf* sa = (const v8bf*)(A + (size_t)(m0 + lrow) * K + k0 + lcol);
      *(v8bf*)&As[lrow][lcol]     = sa[0];
      *(v8bf*)&As[lrow][lcol + 8] = sa[1];
      const v8bf* sb = (const v8bf*)(W + (size_t)(n0 + lrow) * K + k0 + lcol);
      *(v8bf*)&Bs[lrow][lcol]     = sb[0];
      *(v8bf*)&Bs[lrow][lcol + 8] = sb[1];
    }
    __syncthreads();

    v16bf a0 = make_a(&As[32 * wy + nl][0],      kBase);
    v16bf a1 = make_a(&As[32 * wy + 16 + nl][0], kBase);
    v16bf b0 = make_b(&Bs[32 * wx + nl][kOff]);
    v16bf b1 = make_b(&Bs[32 * wx + 16 + nl][kOff]);
    acc[0][0] = wmma_bf16(a0, b0, acc[0][0]);
    acc[0][1] = wmma_bf16(a0, b1, acc[0][1]);
    acc[1][0] = wmma_bf16(a1, b0, acc[1][0]);
    acc[1][1] = wmma_bf16(a1, b1, acc[1][1]);
  }

  // Epilogue. C/D layout: VGPR r, lanes 0-15 -> M=r, lanes 16-31 -> M=r+8; N=lane&15.
#pragma unroll
  for (int mt = 0; mt < 2; ++mt)
#pragma unroll
    for (int nt = 0; nt < 2; ++nt) {
      const int n_g = n0 + 32 * wx + 16 * nt + nl;
      const float bv = bias[n_g];
#pragma unroll
      for (int r = 0; r < 8; ++r) {
        const int m_g = m0 + 32 * wy + 16 * mt + r + 8 * half;
        const float val = (acc[mt][nt][r] + bv) * scale;
        if (mode == 0) {
          const int b = m_g >> 11, s = m_g & (SEQ - 1);
          const int h = n_g >> 6,  d = n_g & (HD - 1);
          ((bf16_t*)out)[(((size_t)(b * NH + h) * SEQ) + s) * HD + d] = (bf16_t)val;
        } else {
          ((float*)out)[(size_t)m_g * N + n_g] = val;
        }
      }
    }
}

// ----- Flash-style sliding-window attention --------------------------------
// Grid: x = SEQ/64 query blocks, y = B*NH.  128 threads (4 waves), wave w owns
// queries q0+16w .. q0+16w+15.  Streams the 320-key union window in 32-key
// chunks through LDS; online softmax with -1e30 masking.

__global__ void __launch_bounds__(128)
attn_sw(const bf16_t* __restrict__ qb, const bf16_t* __restrict__ kb,
        const bf16_t* __restrict__ vb, bf16_t* __restrict__ yb) {
  __shared__ bf16_t Qs[64][72];        // 64 queries x 64 d, rows 144B
  __shared__ bf16_t Ks[32][72];        // 32 keys x 64 d
  __shared__ bf16_t Vt[64][40];        // transposed: [d][key], rows 80B
  __shared__ bf16_t Ps[4][16][32];     // per-wave P tile (C-layout -> A-layout)

  const int tid  = threadIdx.x;
  const int w    = tid >> 5;
  const int lane = tid & 31;
  const int nl   = lane & 15;
  const int half = lane >> 4;
  const int kBase = half * 8;
  const int kOff  = half * 16;

  const int bh = blockIdx.y;                 // b*NH + h
  const int q0 = blockIdx.x * 64;
  const int qw = q0 + 16 * w;
  const size_t base = (size_t)bh * SEQ * HD;

  // Stage Q block (already bias+scale'd by the projection).
  {
    const int row = tid >> 1;
    const int c0  = (tid & 1) * 32;
    const v8bf* s = (const v8bf*)(qb + base + (size_t)(q0 + row) * HD + c0);
    v8bf* d = (v8bf*)&Qs[row][c0];
    d[0] = s[0]; d[1] = s[1]; d[2] = s[2]; d[3] = s[3];
  }

  v8f O[4];
  float mrow[8], lrow[8];
#pragma unroll
  for (int r = 0; r < 8; ++r) { mrow[r] = -1e30f; lrow[r] = 0.f; }
#pragma unroll
  for (int t = 0; t < 4; ++t)
#pragma unroll
    for (int r = 0; r < 8; ++r) O[t][r] = 0.f;

  for (int c = 0; c < 10; ++c) {
    const int keyBase = q0 - WIN + 32 * c;
    __syncthreads();
    // Stage K (row-major) and V (transposed), zero-fill key < 0.
    {
      const int kk = tid >> 2;           // 0..31
      const int d0 = (tid & 3) * 16;     // 0,16,32,48
      const int key = keyBase + kk;
      v8bf zl, zh;
#pragma unroll
      for (int i = 0; i < 8; ++i) { zl[i] = (bf16_t)0.f; zh[i] = (bf16_t)0.f; }
      v8bf klo = zl, khi = zh, vlo = zl, vhi = zh;
      if (key >= 0) {
        const v8bf* sk = (const v8bf*)(kb + base + (size_t)key * HD + d0);
        klo = sk[0]; khi = sk[1];
        const v8bf* sv = (const v8bf*)(vb + base + (size_t)key * HD + d0);
        vlo = sv[0]; vhi = sv[1];
      }
      *(v8bf*)&Ks[kk][d0]     = klo;
      *(v8bf*)&Ks[kk][d0 + 8] = khi;
#pragma unroll
      for (int i = 0; i < 8; ++i) { Vt[d0 + i][kk] = vlo[i]; Vt[d0 + 8 + i][kk] = vhi[i]; }
    }
    __syncthreads();

    // S = Q K^T : two 16x16 score tiles, K-dim 64 = 2 WMMAs each.
    v8f s0, s1;
#pragma unroll
    for (int r = 0; r < 8; ++r) { s0[r] = 0.f; s1[r] = 0.f; }
#pragma unroll
    for (int kc = 0; kc < 2; ++kc) {
      v16bf aq = make_a(&Qs[16 * w + nl][32 * kc], kBase);
      v16bf b0 = make_b(&Ks[nl][32 * kc + kOff]);
      v16bf b1 = make_b(&Ks[16 + nl][32 * kc + kOff]);
      s0 = wmma_bf16(aq, b0, s0);
      s1 = wmma_bf16(aq, b1, s1);
    }

    // Online softmax (per-row stats; rows live across the 16-lane half).
#pragma unroll
    for (int r = 0; r < 8; ++r) {
      const int m_row = r + 8 * half;
      const int qg = qw + m_row;
      const int key0 = keyBase + nl;
      const int key1 = keyBase + 16 + nl;
      const bool v0 = (key0 >= 0) && (key0 <= qg) && (key0 >= qg - WIN);
      const bool v1 = (key1 >= 0) && (key1 <= qg) && (key1 >= qg - WIN);
      float x0 = v0 ? s0[r] : -1e30f;
      float x1 = v1 ? s1[r] : -1e30f;
      float mx = fmaxf(x0, x1);
      mx = fmaxf(mx, __shfl_xor(mx, 1));
      mx = fmaxf(mx, __shfl_xor(mx, 2));
      mx = fmaxf(mx, __shfl_xor(mx, 4));
      mx = fmaxf(mx, __shfl_xor(mx, 8));
      const float mnew = fmaxf(mrow[r], mx);
      const float sc = __expf(mrow[r] - mnew);
      const float p0 = v0 ? __expf(s0[r] - mnew) : 0.f;
      const float p1 = v1 ? __expf(s1[r] - mnew) : 0.f;
      float ps = p0 + p1;
      ps += __shfl_xor(ps, 1);
      ps += __shfl_xor(ps, 2);
      ps += __shfl_xor(ps, 4);
      ps += __shfl_xor(ps, 8);
      lrow[r] = lrow[r] * sc + ps;
      mrow[r] = mnew;
#pragma unroll
      for (int t = 0; t < 4; ++t) O[t][r] *= sc;
      Ps[w][m_row][nl]      = (bf16_t)p0;
      Ps[w][m_row][16 + nl] = (bf16_t)p1;
    }
    // Wave-local LDS RAW: same-wave DS ops are in order; wait them out.
    asm volatile("s_wait_dscnt 0" ::: "memory");

    // O += P @ V : A = P (16x32), B-operand from transposed V.
    v16bf pa = make_a(&Ps[w][nl][0], kBase);
#pragma unroll
    for (int t = 0; t < 4; ++t) {
      v16bf bvv = make_b(&Vt[16 * t + nl][kOff]);
      O[t] = wmma_bf16(pa, bvv, O[t]);
    }
  }

  // Epilogue: y[b][s][h*64+d] = O / l   (bf16, feeds the output projection)
  const int b = bh >> 4;
  const int h = bh & (NH - 1);
#pragma unroll
  for (int t = 0; t < 4; ++t)
#pragma unroll
    for (int r = 0; r < 8; ++r) {
      const int qg = qw + r + 8 * half;
      const int e = h * HD + 16 * t + nl;
      const float val = O[t][r] / lrow[r];
      yb[((size_t)b * SEQ + qg) * EMBED + e] = (bf16_t)val;
    }
}

// ---------------------------------------------------------------------------

extern "C" void kernel_launch(void* const* d_in, const int* in_sizes, int n_in,
                              void* d_out, int out_size, void* d_ws, size_t ws_size,
                              hipStream_t stream) {
  const float* x  = (const float*)d_in[0];
  const float* Wq = (const float*)d_in[1];
  const float* Wk = (const float*)d_in[2];
  const float* Wv = (const float*)d_in[3];
  const float* Wo = (const float*)d_in[4];
  const float* bq = (const float*)d_in[5];
  const float* bk = (const float*)d_in[6];
  const float* bv = (const float*)d_in[7];
  const float* bo = (const float*)d_in[8];
  float* out = (float*)d_out;

  const int M = NB * SEQ;        // 4096
  const int N = EMBED;           // 1024
  const int K = EMBED;           // 1024

  const size_t MiB = 1u << 20;
  char* ws = (char*)d_ws;
  bf16_t* xb  = (bf16_t*)(ws);                 //  8 MiB: x as bf16
  bf16_t* wqb = (bf16_t*)(ws +  8 * MiB);      //  2 MiB each
  bf16_t* wkb = (bf16_t*)(ws + 10 * MiB);
  bf16_t* wvb = (bf16_t*)(ws + 12 * MiB);
  bf16_t* wob = (bf16_t*)(ws + 14 * MiB);
  bf16_t* qbf = (bf16_t*)(ws + 16 * MiB);      //  8 MiB each, [B][H][S][D]
  bf16_t* kbf = (bf16_t*)(ws + 24 * MiB);
  bf16_t* vbf = (bf16_t*)(ws + 32 * MiB);
  bf16_t* ybf = (bf16_t*)(ws + 40 * MiB);      //  8 MiB, [B][S][E]

  const int nX = M * K;          // 4,194,304
  const int nW = N * K;          // 1,048,576
  cvt_f32_bf16<<<(nX / 4 + 255) / 256, 256, 0, stream>>>(x,  xb,  nX / 4);
  cvt_f32_bf16<<<(nW / 4 + 255) / 256, 256, 0, stream>>>(Wq, wqb, nW / 4);
  cvt_f32_bf16<<<(nW / 4 + 255) / 256, 256, 0, stream>>>(Wk, wkb, nW / 4);
  cvt_f32_bf16<<<(nW / 4 + 255) / 256, 256, 0, stream>>>(Wv, wvb, nW / 4);
  cvt_f32_bf16<<<(nW / 4 + 255) / 256, 256, 0, stream>>>(Wo, wob, nW / 4);

  dim3 gg(N / 64, M / 64);       // (16, 64)
  const float qscale = 0.125f;   // HD^-0.5, applied after bias per reference
  gemm_bf16<<<gg, 128, 0, stream>>>(xb, wqb, bq, qbf, M, N, K, 0, qscale);
  gemm_bf16<<<gg, 128, 0, stream>>>(xb, wkb, bk, kbf, M, N, K, 0, 1.0f);
  gemm_bf16<<<gg, 128, 0, stream>>>(xb, wvb, bv, vbf, M, N, K, 0, 1.0f);

  dim3 ga(SEQ / 64, NB * NH);    // (32, 32)
  attn_sw<<<ga, 128, 0, stream>>>(qbf, kbf, vbf, ybf);

  gemm_bf16<<<gg, 128, 0, stream>>>(ybf, wob, bo, out, M, N, K, 1, 1.0f);
}
